// GraphTransformer_20401094656270
// MI455X (gfx1250) — matrix operand order repaired and checked by reading, not compiled, not run on hardware
//
#include <hip/hip_runtime.h>
#include <hip/hip_bf16.h>

typedef __attribute__((ext_vector_type(16))) _Float16 v16h;
typedef __attribute__((ext_vector_type(8)))  float    v8f;

#define FD 128          // feature dim
#define GEMM_ROWS 64    // rows per block (4 waves x 16 rows)

// ---------------------------------------------------------------------------
// order-preserving float <-> uint encoding for atomicMax-based segment max
// ---------------------------------------------------------------------------
__device__ __forceinline__ unsigned encf(float f) {
    unsigned u = __float_as_uint(f);
    return (u & 0x80000000u) ? ~u : (u | 0x80000000u);
}
__device__ __forceinline__ float decf(unsigned u) {
    u = (u & 0x80000000u) ? (u & 0x7FFFFFFFu) : ~u;
    return __uint_as_float(u);
}

// ---------------------------------------------------------------------------
// elementwise helpers
// ---------------------------------------------------------------------------
__global__ void gt_zero_u32(unsigned* __restrict__ p, int n) {
    int i = blockIdx.x * blockDim.x + threadIdx.x;
    if (i < n) p[i] = 0u;
}

__global__ void gt_f32_to_f16(const float* __restrict__ in, _Float16* __restrict__ out, int n) {
    int i = blockIdx.x * blockDim.x + threadIdx.x;
    if (i < n) out[i] = (_Float16)in[i];
}

__global__ void gt_leaky_to_f16(const float* __restrict__ in, _Float16* __restrict__ out, int n) {
    int i = blockIdx.x * blockDim.x + threadIdx.x;
    if (i < n) {
        float y = in[i];
        y = (y >= 0.0f) ? y : 0.01f * y;
        out[i] = (_Float16)y;
    }
}

// ---------------------------------------------------------------------------
// WMMA GEMM: Y[N,128] = H16[N,128](f16) @ W[128,128](f32) + b
// block = 128 threads (4 waves), each block does 64 rows x 128 cols.
// LDS tiles are stored FRAGMENT-MAJOR: each lane's 16-half WMMA fragment is
// contiguous (32B) so fragment loads lower to 2x ds_load_b128.
//
// A fragment (16x32): element e <-> K%32 = e + (e>=8?8:0) + 8*half
//   inverse: g=(K%32)>>3, half=g&1, e=(g>>1)*8 + (K&7)
//   slot(wv,kt,lane) = ((wv*4+kt)*32 + lane)*16 + e
// B fragment (32x16): element e <-> K%32 = e + (e>=8?8:0) + 8*half, col = ct*16 + l15
//   slot(kt,ct,lane) = ((kt*8+ct)*32 + lane)*16 + e
// ---------------------------------------------------------------------------
__global__ __launch_bounds__(128) void gt_gemm_wmma(
    const _Float16* __restrict__ h16,
    const float* __restrict__ W, const float* __restrict__ bias,
    float* __restrict__ Y, int N)
{
    __shared__ _Float16 sA[GEMM_ROWS * FD];   // 16 KB, fragment-major
    __shared__ _Float16 sB[FD * FD];          // 32 KB, fragment-major

    const int tid  = threadIdx.x;
    const int lane = tid & 31;
    const int wv   = tid >> 5;
    const int half = lane >> 4;
    const int l15  = lane & 15;
    const int rowBlock = blockIdx.x * GEMM_ROWS;

    // ---- Stage A tile (64x128 f16) directly into fragment-major layout.
    // Each vec8 chunk (row r, cols c8*8..+7) maps to ONE contiguous 8-half
    // span of a fragment: kt=c8>>2, g=c8&3 -> half=g&1, e0=(g>>1)*8.
#pragma unroll
    for (int i = tid; i < GEMM_ROWS * FD / 8; i += 128) {
        int r  = i >> 4;
        int c8 = i & 15;
        int grow = rowBlock + r;
        uint4 val = {0u, 0u, 0u, 0u};
        if (grow < N) val = *(const uint4*)(h16 + grow * FD + c8 * 8);
        int kt   = c8 >> 2;
        int g    = c8 & 3;
        int hf   = g & 1;
        int e0   = (g >> 1) * 8;
        int ln   = hf * 16 + (r & 15);
        int wvr  = r >> 4;
        _Float16* dst = sA + ((wvr * 4 + kt) * 32 + ln) * 16 + e0;
        *(uint4*)dst = val;
    }

    // ---- Stage W (f32 -> f16) into fragment-major layout.
    // Coalesced float4 reads of W rows; 4 scattered b16 stores each.
    for (int i = tid; i < FD * FD / 4; i += 128) {
        int k  = (i * 4) >> 7;         // row of W
        int j0 = (i * 4) & 127;        // first of 4 consecutive cols
        float4 f = *(const float4*)(W + i * 4);
        int kt = k >> 5;
        int kk = k & 31;
        int hf = (kk >> 3) & 1;
        int e  = (kk & 7) + ((kk >> 4) << 3);
        float fv[4] = {f.x, f.y, f.z, f.w};
#pragma unroll
        for (int t = 0; t < 4; ++t) {
            int j  = j0 + t;
            int ct = j >> 4;
            int ln = hf * 16 + (j & 15);
            sB[((kt * 8 + ct) * 32 + ln) * 16 + e] = (_Float16)fv[t];
        }
    }
    __syncthreads();

    v8f acc[8];
#pragma unroll
    for (int ct = 0; ct < 8; ++ct) acc[ct] = (v8f){0,0,0,0,0,0,0,0};

    const _Float16* aBase = sA + (wv * 4 * 32 + lane) * 16;  // + kt*512
    const _Float16* bBase = sB + lane * 16;                  // + (kt*8+ct)*512

#pragma unroll
    for (int kt = 0; kt < 4; ++kt) {
        v16h a = *(const v16h*)(aBase + kt * 512);
#pragma unroll
        for (int ct = 0; ct < 8; ++ct) {
            v16h b = *(const v16h*)(bBase + (kt * 8 + ct) * 512);
            acc[ct] = __builtin_amdgcn_wmma_f32_16x16x32_f16(
                false, a, false, b, (short)0, acc[ct], false, false);
        }
    }

    // Store with bias; D tile: row = r + 8*half, col = l15
#pragma unroll
    for (int ct = 0; ct < 8; ++ct) {
        int col = ct * 16 + l15;
        float bb = bias[col];
#pragma unroll
        for (int r = 0; r < 8; ++r) {
            int row = rowBlock + wv * 16 + half * 8 + r;
            if (row < N) Y[row * FD + col] = acc[ct][r] + bb;
        }
    }
}

// ---------------------------------------------------------------------------
// Edge phase: one wave (32 lanes x float4) per edge
// ---------------------------------------------------------------------------
__global__ __launch_bounds__(256) void gt_edge_score(
    const float* __restrict__ q, const float* __restrict__ k,
    const int* __restrict__ src, const int* __restrict__ dst,
    float* __restrict__ score, unsigned* __restrict__ mEnc, int E)
{
    int gid  = blockIdx.x * 256 + threadIdx.x;
    int e    = gid >> 5;
    int lane = threadIdx.x & 31;
    if (e >= E) return;
    int s = src[e], d = dst[e];
    float4 a = *(const float4*)(q + (size_t)d * FD + lane * 4);
    float4 b = *(const float4*)(k + (size_t)s * FD + lane * 4);
    float p = a.x * b.x + a.y * b.y + a.z * b.z + a.w * b.w;
#pragma unroll
    for (int off = 16; off > 0; off >>= 1) p += __shfl_xor(p, off, 32);
    if (lane == 0) {
        float sc = p * 0.08838834764831845f;   // 1/sqrt(128)
        score[e] = sc;
        atomicMax(mEnc + d, encf(sc));
    }
}

__global__ void gt_edge_exp(
    float* __restrict__ score, const unsigned* __restrict__ mEnc,
    const int* __restrict__ dst, float* __restrict__ denom, int E)
{
    int e = blockIdx.x * blockDim.x + threadIdx.x;
    if (e >= E) return;
    int d = dst[e];
    float ev = __expf(score[e] - decf(mEnc[d]));
    score[e] = ev;
    atomicAdd(denom + d, ev);
}

__global__ __launch_bounds__(256) void gt_edge_agg(
    const float* __restrict__ evals, const float* __restrict__ denom,
    const float* __restrict__ v, const int* __restrict__ src,
    const int* __restrict__ dst, float* __restrict__ hnext, int E)
{
    int gid  = blockIdx.x * 256 + threadIdx.x;
    int e    = gid >> 5;
    int lane = threadIdx.x & 31;
    if (e >= E) return;
    int s = src[e], d = dst[e];
    float alpha = evals[e] / denom[d];
    float4 vv = *(const float4*)(v + (size_t)s * FD + lane * 4);
    float* out = hnext + (size_t)d * FD + lane * 4;
    atomicAdd(out + 0, alpha * vv.x);
    atomicAdd(out + 1, alpha * vv.y);
    atomicAdd(out + 2, alpha * vv.z);
    atomicAdd(out + 3, alpha * vv.w);
}

// ---------------------------------------------------------------------------
// Pooling
// ---------------------------------------------------------------------------
__global__ void gt_pool_count(const int* __restrict__ batch, float* __restrict__ counts, int N) {
    int n = blockIdx.x * blockDim.x + threadIdx.x;
    if (n < N) atomicAdd(counts + batch[n], 1.0f);
}

__global__ __launch_bounds__(256) void gt_pool_sum(
    const float* __restrict__ node, const int* __restrict__ batch,
    float* __restrict__ gsum, int N)
{
    int gid  = blockIdx.x * 256 + threadIdx.x;
    int n    = gid >> 5;
    int lane = threadIdx.x & 31;
    if (n >= N) return;
    int b = batch[n];
    float4 x = *(const float4*)(node + (size_t)n * FD + lane * 4);
    float* o = gsum + (size_t)b * FD + lane * 4;
    atomicAdd(o + 0, x.x);
    atomicAdd(o + 1, x.y);
    atomicAdd(o + 2, x.z);
    atomicAdd(o + 3, x.w);
}

__global__ void gt_pool_div(const float* __restrict__ gsum, const float* __restrict__ counts,
                            float* __restrict__ out, int BD) {
    int i = blockIdx.x * blockDim.x + threadIdx.x;
    if (i < BD) out[i] = gsum[i] / fmaxf(counts[i >> 7], 1.0f);
}

// ---------------------------------------------------------------------------
// launcher
// ---------------------------------------------------------------------------
extern "C" void kernel_launch(void* const* d_in, const int* in_sizes, int n_in,
                              void* d_out, int out_size, void* d_ws, size_t ws_size,
                              hipStream_t stream) {
    const float* x     = (const float*)d_in[0];
    const int*   edge  = (const int*)d_in[1];
    const int*   batch = (const int*)d_in[2];
    const float* Wq = (const float*)d_in[3];
    const float* bq = (const float*)d_in[4];
    const float* Wk = (const float*)d_in[5];
    const float* bk = (const float*)d_in[6];
    const float* Wv = (const float*)d_in[7];
    const float* bv = (const float*)d_in[8];
    const float* Ws = (const float*)d_in[9];
    const float* bs = (const float*)d_in[10];

    const int N  = in_sizes[0] / FD;
    const int E  = in_sizes[1] / 2;
    const int Bg = (out_size - N * FD) / FD;      // number of graphs
    const int L  = in_sizes[4] / FD;              // 3
    const size_t ND = (size_t)N * FD;

    const int* src = edge;          // edge_index[0]
    const int* dst = edge + E;      // edge_index[1]

    // workspace carve (all 16B-aligned for this problem's sizes)
    char* ws = (char*)d_ws;
    float*     q     = (float*)ws;            ws += ND * sizeof(float);
    float*     kbuf  = (float*)ws;            ws += ND * sizeof(float);
    float*     vbuf  = (float*)ws;            ws += ND * sizeof(float);
    float*     hnext = (float*)ws;            ws += ND * sizeof(float);
    _Float16*  h16   = (_Float16*)ws;         ws += ND * sizeof(_Float16);
    float*     evals = (float*)ws;            ws += (size_t)E * sizeof(float);
    unsigned*  mEnc  = (unsigned*)ws;         ws += (size_t)N * sizeof(unsigned);
    float*     denom = (float*)ws;            ws += (size_t)N * sizeof(float);
    float*     gsum  = (float*)ws;            ws += (size_t)Bg * FD * sizeof(float);
    float*     counts= (float*)ws;            ws += (size_t)Bg * sizeof(float);
    (void)ws_size; (void)n_in;

    float* outNode  = (float*)d_out;
    float* outGraph = (float*)d_out + ND;

    const int T = 256;
    const int gemmBlocks = (N + GEMM_ROWS - 1) / GEMM_ROWS;
    const int edgeWaveBlocks = (E * 32 + T - 1) / T;

    // layer 0 input: x -> f16
    gt_f32_to_f16<<<(int)((ND + T - 1) / T), T, 0, stream>>>(x, h16, (int)ND);

    for (int l = 0; l < L; ++l) {
        const size_t wo = (size_t)l * FD * FD;
        const size_t bo = (size_t)l * FD;
        float* skipOut = (l == L - 1) ? outNode : hnext;

        gt_gemm_wmma<<<gemmBlocks, 128, 0, stream>>>(h16, Wq + wo, bq + bo, q,     N);
        gt_gemm_wmma<<<gemmBlocks, 128, 0, stream>>>(h16, Wk + wo, bk + bo, kbuf,  N);
        gt_gemm_wmma<<<gemmBlocks, 128, 0, stream>>>(h16, Wv + wo, bv + bo, vbuf,  N);
        gt_gemm_wmma<<<gemmBlocks, 128, 0, stream>>>(h16, Ws + wo, bs + bo, skipOut, N);

        // zero segment-max encodings and denominators (adjacent buffers)
        gt_zero_u32<<<(2 * N + T - 1) / T, T, 0, stream>>>(mEnc, 2 * N);

        gt_edge_score<<<edgeWaveBlocks, T, 0, stream>>>(q, kbuf, src, dst, evals, mEnc, E);
        gt_edge_exp<<<(E + T - 1) / T, T, 0, stream>>>(evals, mEnc, dst, denom, E);
        gt_edge_agg<<<edgeWaveBlocks, T, 0, stream>>>(evals, denom, vbuf, src, dst, skipOut, E);

        if (l < L - 1) {
            // leaky_relu fused with fp16 conversion for next layer's GEMM input
            gt_leaky_to_f16<<<(int)((ND + T - 1) / T), T, 0, stream>>>(hnext, h16, (int)ND);
        }
    }

    // global mean pool (gsum and counts are adjacent -> one zero pass)
    gt_zero_u32<<<(Bg * FD + Bg + T - 1) / T, T, 0, stream>>>((unsigned*)gsum, Bg * FD + Bg);
    gt_pool_count<<<(N + T - 1) / T, T, 0, stream>>>(batch, counts, N);
    gt_pool_sum<<<(N * 32 + T - 1) / T, T, 0, stream>>>(outNode, batch, gsum, N);
    gt_pool_div<<<(Bg * FD + T - 1) / T, T, 0, stream>>>(gsum, counts, outGraph, Bg * FD);
}
